// AWD_LSTM_87016037417057
// MI455X (gfx1250) — matrix operand order repaired and checked by reading, not compiled, not run on hardware
//
#include <hip/hip_runtime.h>
#include <hip/hip_bf16.h>

typedef __attribute__((ext_vector_type(16))) __bf16 v16bf;
typedef __attribute__((ext_vector_type(8)))  float  v8f;
typedef __bf16 bf16;

#define kHID  1150
#define kHP   1152
#define kG4   4608
#define kEMB  400
#define kEP   416
#define kB    80
#define kS    70
#define kR    5600            // kS*kB
#define kNTOK 33278
#define kNDP  33280
#define KT_E  13              // 416/32
#define KT_H  36              // 1152/32
#define NT_G  288             // 4608/16
#define NT_D  2080            // 33280/16
#define RT    350             // 5600/16
#define MT_B  5               // 80/16
#define TILE  ((size_t)512)   // bf16 elements per packed 16x32 / 32x16 tile
#define DTILE ((size_t)256)   // f32 elements per packed 16x16 D tile

// ---- WMMA register layouts (CDNA5 ISA 7.12.2, wave32) ----
// A (16x32 bf16): lanes 0-15 hold M, elems 0-7 = K0-7, elems 8-15 = K16-23;
//                 lanes 16-31 same M, K8-15 / K24-31.
static __device__ __forceinline__ int a_idx(int m, int kk) {
  int grp  = (kk & 15) >> 3;
  int lane = m + (grp << 4);
  int elem = (kk & 7) + ((kk >> 4) << 3);
  return lane * 16 + elem;
}
// B (32x16 bf16): packed via a_idx like A, N in place of M (lane = N + 16*((K>>3)&1))
// D (16x16 f32): lane = (n&15) + 16*(m/8), vgpr = m%8  -> packed as lane*8+v

static __device__ __forceinline__ v8f wmma_bf16(v16bf a, v16bf b, v8f c) {
  return __builtin_amdgcn_wmma_f32_16x16x32_bf16(false, a, false, b, (short)0, c,
                                                 false, false);
}

// Software-pipelined 4-accumulator GEMM step: one A tile stream, 4 B tile streams
// at constant element stride BSTRIDE (folds to immediate offsets). Next-iteration
// fragments are loaded before the current WMMAs so loads overlap matrix ops.
template <int KT, size_t BSTRIDE>
static __device__ __forceinline__ void gemm4(const bf16* __restrict__ aBase,
                                             const bf16* __restrict__ bBase,
                                             int lane, v8f* acc) {
  const bf16* aP = aBase + (size_t)lane * 16;
  const bf16* bP = bBase + (size_t)lane * 16;
  v16bf a  = *(const v16bf*)aP;
  v16bf b0 = *(const v16bf*)bP;
  v16bf b1 = *(const v16bf*)(bP + BSTRIDE);
  v16bf b2 = *(const v16bf*)(bP + 2 * BSTRIDE);
  v16bf b3 = *(const v16bf*)(bP + 3 * BSTRIDE);
#pragma unroll 1
  for (int kt = 0; kt < KT - 1; ++kt) {
    aP += TILE;
    bP += TILE;
    v16bf an  = *(const v16bf*)aP;
    v16bf bn0 = *(const v16bf*)bP;
    v16bf bn1 = *(const v16bf*)(bP + BSTRIDE);
    v16bf bn2 = *(const v16bf*)(bP + 2 * BSTRIDE);
    v16bf bn3 = *(const v16bf*)(bP + 3 * BSTRIDE);
    acc[0] = wmma_bf16(a, b0, acc[0]);
    acc[1] = wmma_bf16(a, b1, acc[1]);
    acc[2] = wmma_bf16(a, b2, acc[2]);
    acc[3] = wmma_bf16(a, b3, acc[3]);
    a = an; b0 = bn0; b1 = bn1; b2 = bn2; b3 = bn3;
  }
  acc[0] = wmma_bf16(a, b0, acc[0]);
  acc[1] = wmma_bf16(a, b1, acc[1]);
  acc[2] = wmma_bf16(a, b2, acc[2]);
  acc[3] = wmma_bf16(a, b3, acc[3]);
}

// ===================== packing kernels =====================
__global__ void k_pack_gate(const float* __restrict__ src, bf16* __restrict__ dst,
                            int K, int KT) {
  int Kp = KT * 32;
  long long tot = (long long)kG4 * Kp;
  long long idx = (long long)blockIdx.x * blockDim.x + threadIdx.x;
  if (idx >= tot) return;
  int n = (int)(idx / Kp), k = (int)(idx % Kp);
  int q = n / kHP, j = n % kHP;
  float v = 0.f;
  if (j < kHID && k < K) v = src[(size_t)(q * kHID + j) * K + k];
  dst[(size_t)((n >> 4) * KT + (k >> 5)) * TILE + (size_t)a_idx(n & 15, k & 31)] = (bf16)v;
}

__global__ void k_pack_dec(const float* __restrict__ src, bf16* __restrict__ dst) {
  long long tot = (long long)kNDP * kHP;
  long long idx = (long long)blockIdx.x * blockDim.x + threadIdx.x;
  if (idx >= tot) return;
  int n = (int)(idx / kHP), k = (int)(idx % kHP);
  float v = 0.f;
  if (n < kNTOK && k < kHID) v = src[(size_t)n * kHID + k];
  dst[(size_t)((n >> 4) * KT_H + (k >> 5)) * TILE + (size_t)a_idx(n & 15, k & 31)] = (bf16)v;
}

__global__ void k_embed(const int* __restrict__ x, const float* __restrict__ emb,
                        bf16* __restrict__ Xep) {
  long long tot = (long long)kR * kEP;
  long long idx = (long long)blockIdx.x * blockDim.x + threadIdx.x;
  if (idx >= tot) return;
  int r = (int)(idx / kEP), k = (int)(idx % kEP);
  float v = 0.f;
  if (k < kEMB) v = emb[(size_t)x[r] * kEMB + k];
  Xep[(size_t)((r >> 4) * KT_E + (k >> 5)) * TILE + a_idx(r & 15, k & 31)] = (bf16)v;
}

__global__ void k_init(const float* __restrict__ h0, const float* __restrict__ c0,
                       float* __restrict__ Hf, float* __restrict__ Cf,
                       bf16* __restrict__ Hb0, bf16* __restrict__ Hb1, bf16* __restrict__ Hb2) {
  int tot = 3 * kB * kHP;
  int idx = blockIdx.x * blockDim.x + threadIdx.x;
  if (idx >= tot) return;
  int l = idx / (kB * kHP);
  int rem = idx % (kB * kHP);
  int b = rem / kHP, j = rem % kHP;
  float h = 0.f, c = 0.f;
  if (j < kHID) {
    h = h0[((size_t)l * kB + b) * kHID + j];
    c = c0[((size_t)l * kB + b) * kHID + j];
  }
  Hf[idx] = h;
  Cf[idx] = c;
  bf16* Hb = (l == 0) ? Hb0 : ((l == 1) ? Hb1 : Hb2);
  Hb[(size_t)((b >> 4) * KT_H + (j >> 5)) * TILE + a_idx(b & 15, j & 31)] = (bf16)h;
}

// ===================== WMMA GEMM kernels =====================
// P1 tiles (D-layout, 256 f32/tile): tile (rt*NT_G + nt), element lane*8+v.
__global__ void k_gemm_p1(const bf16* __restrict__ Xep, const bf16* __restrict__ W,
                          const float* __restrict__ bi, const float* __restrict__ bh,
                          float* __restrict__ P1p) {
  int wv = blockIdx.x * (blockDim.x >> 5) + (threadIdx.x >> 5);
  int lane = threadIdx.x & 31;
  int mt = wv / 72, ng = wv % 72;      // 350 row tiles x 72 groups of 4 n-tiles
  v8f acc[4] = {};
  gemm4<KT_E, (size_t)KT_E * TILE>(Xep + (size_t)mt * KT_E * TILE,
                                   W + (size_t)(ng * 4) * KT_E * TILE, lane, acc);
  int nl = lane & 15;
  int q = ng / 18, jg = ng % 18;       // group of 4 tiles never straddles a gate
#pragma unroll
  for (int i = 0; i < 4; ++i) {
    int jt = jg * 4 + i;
    int j  = jt * 16 + nl;
    float bias = (j < kHID) ? (bi[q * kHID + j] + bh[q * kHID + j]) : 0.f;
    v8f r = acc[i];
#pragma unroll
    for (int v = 0; v < 8; ++v) r[v] += bias;
    int nt = q * 72 + jt;
    *(v8f*)(P1p + (size_t)(mt * NT_G + nt) * DTILE + (size_t)lane * 8) = r;
  }
}

// fused LSTM cell
__global__ void k_cell(const float* __restrict__ P1p,  // D-layout preacts or null
                       const bf16* __restrict__ Xp, const bf16* __restrict__ Wxp,
                       const bf16* __restrict__ Hbp, const bf16* __restrict__ Whp,
                       const float* __restrict__ b_ih, const float* __restrict__ b_hh,
                       float* __restrict__ Cst, float* __restrict__ Hst,
                       bf16* __restrict__ Hout, bf16* __restrict__ Zsp, int t) {
  int wv = blockIdx.x * (blockDim.x >> 5) + (threadIdx.x >> 5);
  int lane = threadIdx.x & 31;
  int mt = wv / 72, jt = wv % 72;      // 5 batch tiles x 72 hidden tiles
  int m0 = (lane >> 4) * 8, nl = lane & 15;
  v8f acc[4] = {};

  if (P1p) {  // layer 1: input-path preactivations (biases folded) via v8f loads
    const float* pt = P1p + (size_t)(t * MT_B + mt) * NT_G * DTILE + (size_t)lane * 8;
#pragma unroll
    for (int q = 0; q < 4; ++q)
      acc[q] = *(const v8f*)(pt + (size_t)(q * 72 + jt) * DTILE);
  } else {    // layers 2/3: x-input GEMM against lower layer's fresh h
    gemm4<KT_H, (size_t)(72 * KT_H) * TILE>(Xp + (size_t)mt * KT_H * TILE,
                                            Wxp + (size_t)jt * KT_H * TILE, lane, acc);
  }
  // recurrent GEMM
  gemm4<KT_H, (size_t)(72 * KT_H) * TILE>(Hbp + (size_t)mt * KT_H * TILE,
                                          Whp + (size_t)jt * KT_H * TILE, lane, acc);
  // gates + state update
#pragma unroll
  for (int v = 0; v < 8; ++v) {
    int bb = mt * 16 + m0 + v;
    int j  = jt * 16 + nl;
    float pi = acc[0][v], pf = acc[1][v], po = acc[2][v], pg = acc[3][v];
    if (b_ih && j < kHID) {
      pi += b_ih[j]            + b_hh[j];
      pf += b_ih[kHID + j]     + b_hh[kHID + j];
      po += b_ih[2 * kHID + j] + b_hh[2 * kHID + j];
      pg += b_ih[3 * kHID + j] + b_hh[3 * kHID + j];
    }
    float ig = 1.f / (1.f + __expf(-pi));
    float fg = 1.f / (1.f + __expf(-pf));
    float og = 1.f / (1.f + __expf(-po));
    float gg = tanhf(pg);
    size_t sidx = (size_t)bb * kHP + j;
    float cp = Cst[sidx];
    float ct = fg * cp + ig * gg;
    float ht = og * tanhf(ct);
    if (j >= kHID) { ct = 0.f; ht = 0.f; }      // keep padding inert
    Cst[sidx] = ct;
    Hst[sidx] = ht;
    Hout[(size_t)((bb >> 4) * KT_H + (j >> 5)) * TILE + a_idx(bb & 15, j & 31)] = (bf16)ht;
    if (Zsp) {
      int r = t * kB + bb;                      // 80 = 5*16 -> tiles align
      Zsp[(size_t)((r >> 4) * KT_H + (j >> 5)) * TILE + a_idx(r & 15, j & 31)] = (bf16)ht;
    }
  }
}

// decoded = zs @ w_dec^T + b_dec : [5600, 33278] fp32
__global__ void k_dec(const bf16* __restrict__ Zsp, const bf16* __restrict__ Wd,
                      const float* __restrict__ bd, float* __restrict__ out) {
  int wv = blockIdx.x * (blockDim.x >> 5) + (threadIdx.x >> 5);
  int lane = threadIdx.x & 31;
  int mt = wv / 520, ng = wv % 520;    // 350 row tiles x 520 groups of 4 n-tiles
  v8f acc[4] = {};
  gemm4<KT_H, (size_t)KT_H * TILE>(Zsp + (size_t)mt * KT_H * TILE,
                                   Wd + (size_t)(ng * 4) * KT_H * TILE, lane, acc);
  int m0 = (lane >> 4) * 8, nl = lane & 15;
#pragma unroll
  for (int i = 0; i < 4; ++i) {
    int n = (ng * 4 + i) * 16 + nl;
    if (n < kNTOK) {
      float bias = bd[n];
#pragma unroll
      for (int v = 0; v < 8; ++v) {
        int r = mt * 16 + m0 + v;
        out[(size_t)r * kNTOK + n] = acc[i][v] + bias;
      }
    }
  }
}

__global__ void k_final(const float* __restrict__ Hf, const float* __restrict__ Cf,
                        float* __restrict__ outH, float* __restrict__ outC) {
  int tot = 3 * kB * kHID;
  int idx = blockIdx.x * blockDim.x + threadIdx.x;
  if (idx >= tot) return;
  int l = idx / (kB * kHID);
  int rem = idx % (kB * kHID);
  int b = rem / kHID, j = rem % kHID;
  size_t s = ((size_t)l * kB + b) * kHP + j;
  outH[idx] = Hf[s];
  outC[idx] = Cf[s];
}

// ===================== host =====================
extern "C" void kernel_launch(void* const* d_in, const int* in_sizes, int n_in,
                              void* d_out, int out_size, void* d_ws, size_t ws_size,
                              hipStream_t stream) {
  (void)in_sizes; (void)n_in; (void)out_size; (void)ws_size;
  const int*   x   = (const int*)d_in[0];
  const float* h0  = (const float*)d_in[1];
  const float* c0  = (const float*)d_in[2];
  const float* emb = (const float*)d_in[3];
  const float* wih[3] = {(const float*)d_in[4],  (const float*)d_in[8],  (const float*)d_in[12]};
  const float* bih[3] = {(const float*)d_in[5],  (const float*)d_in[9],  (const float*)d_in[13]};
  const float* whh[3] = {(const float*)d_in[6],  (const float*)d_in[10], (const float*)d_in[14]};
  const float* bhh[3] = {(const float*)d_in[7],  (const float*)d_in[11], (const float*)d_in[15]};
  const float* wdec = (const float*)d_in[16];
  const float* bdec = (const float*)d_in[17];
  float* out = (float*)d_out;

  // ---- workspace carve (256B aligned regions) ----
  uint8_t* p = (uint8_t*)d_ws;
  auto carve = [&](size_t bytes) -> void* {
    void* r = (void*)p;
    p += (bytes + 255) & ~(size_t)255;
    return r;
  };
  bf16* Wxp[3]; bf16* Whp[3];
  Wxp[0] = (bf16*)carve((size_t)NT_G * KT_E * TILE * 2);
  Whp[0] = (bf16*)carve((size_t)NT_G * KT_H * TILE * 2);
  Wxp[1] = (bf16*)carve((size_t)NT_G * KT_H * TILE * 2);
  Whp[1] = (bf16*)carve((size_t)NT_G * KT_H * TILE * 2);
  Wxp[2] = (bf16*)carve((size_t)NT_G * KT_H * TILE * 2);
  Whp[2] = (bf16*)carve((size_t)NT_G * KT_H * TILE * 2);
  bf16*  Wdp = (bf16*)carve((size_t)NT_D * KT_H * TILE * 2);
  bf16*  Xep = (bf16*)carve((size_t)RT * KT_E * TILE * 2);
  float* P1p = (float*)carve((size_t)RT * NT_G * DTILE * 4);
  bf16*  Zsp = (bf16*)carve((size_t)RT * KT_H * TILE * 2);
  bf16*  Hb[3][2];
  for (int l = 0; l < 3; ++l)
    for (int s = 0; s < 2; ++s)
      Hb[l][s] = (bf16*)carve((size_t)MT_B * KT_H * TILE * 2);
  float* Hf = (float*)carve((size_t)3 * kB * kHP * 4);
  float* Cf = (float*)carve((size_t)3 * kB * kHP * 4);

  // ---- pack weights to bf16 WMMA tiles (fits in 192MB L2) ----
  {
    int ge = kG4 * kEP, gh = kG4 * kHP;
    k_pack_gate<<<(ge + 255) / 256, 256, 0, stream>>>(wih[0], Wxp[0], kEMB, KT_E);
    k_pack_gate<<<(gh + 255) / 256, 256, 0, stream>>>(whh[0], Whp[0], kHID, KT_H);
    k_pack_gate<<<(gh + 255) / 256, 256, 0, stream>>>(wih[1], Wxp[1], kHID, KT_H);
    k_pack_gate<<<(gh + 255) / 256, 256, 0, stream>>>(whh[1], Whp[1], kHID, KT_H);
    k_pack_gate<<<(gh + 255) / 256, 256, 0, stream>>>(wih[2], Wxp[2], kHID, KT_H);
    k_pack_gate<<<(gh + 255) / 256, 256, 0, stream>>>(whh[2], Whp[2], kHID, KT_H);
    long long de = (long long)kNDP * kHP;
    k_pack_dec<<<(unsigned)((de + 255) / 256), 256, 0, stream>>>(wdec, Wdp);
  }
  // ---- embedding gather + state init ----
  k_embed<<<(kR * kEP + 255) / 256, 256, 0, stream>>>(x, emb, Xep);
  k_init<<<(3 * kB * kHP + 255) / 256, 256, 0, stream>>>(h0, c0, Hf, Cf,
                                                         Hb[0][0], Hb[1][0], Hb[2][0]);
  // ---- hoist layer-1 input GEMM (all timesteps, parallel) ----
  k_gemm_p1<<<(RT * 72) / 4, 128, 0, stream>>>(Xep, Wxp[0], bih[0], bhh[0], P1p);

  // ---- sequential scan: 70 steps x 3 fused cells ----
  for (int t = 0; t < kS; ++t) {
    int rb = t & 1, wb = 1 - rb;  // ping-pong h buffers
    k_cell<<<90, 128, 0, stream>>>(P1p, nullptr, nullptr,
                                   Hb[0][rb], Whp[0], nullptr, nullptr,
                                   Cf + 0 * kB * kHP, Hf + 0 * kB * kHP,
                                   Hb[0][wb], nullptr, t);
    k_cell<<<90, 128, 0, stream>>>(nullptr, Hb[0][wb], Wxp[1],
                                   Hb[1][rb], Whp[1], bih[1], bhh[1],
                                   Cf + 1 * kB * kHP, Hf + 1 * kB * kHP,
                                   Hb[1][wb], nullptr, t);
    k_cell<<<90, 128, 0, stream>>>(nullptr, Hb[1][wb], Wxp[2],
                                   Hb[2][rb], Whp[2], bih[2], bhh[2],
                                   Cf + 2 * kB * kHP, Hf + 2 * kB * kHP,
                                   Hb[2][wb], Zsp, t);
  }

  // ---- decoder GEMM (compute-bound, bf16 WMMA) ----
  k_dec<<<(RT * 520) / 4, 128, 0, stream>>>(Zsp, Wdp, bdec, out);

  // ---- final h/c outputs ----
  float* outH = out + (size_t)kR * kNTOK;
  float* outC = outH + (size_t)3 * kB * kHID;
  k_final<<<(3 * kB * kHID + 255) / 256, 256, 0, stream>>>(Hf, Cf, outH, outC);
}
